// ScaledDotProductAttention_87119116632421
// MI455X (gfx1250) — compile-verified, matrix operand order checked by reading, not run on hardware
//
#include <hip/hip_runtime.h>

typedef __attribute__((ext_vector_type(16))) _Float16 v16h;
typedef __attribute__((ext_vector_type(8)))  float    v8f;

#define B_  32
#define L_  2048
#define D_  128
#define BM  32                       // Q rows per workgroup
#define NWAVES 8
#define SCALE 0.0883883476483184405f // 1/sqrt(128)

// K-dimension packing for 16-bit WMMA fragments (ISA 7.12.2):
// fragment dword v (0..7) holds K pair {koff, koff+1},
// koff = (v<4 ? 2v : 8+2v) + (lane>=16 ? 8 : 0)
__device__ __forceinline__ int kOff(int v, int hi) {
    return ((v < 4) ? 2 * v : 8 + 2 * v) + (hi ? 8 : 0);
}

__global__ __launch_bounds__(256) void
sdpa_wmma_kernel(const float* __restrict__ qu,
                 const float* __restrict__ ke,
                 const float* __restrict__ va,
                 float* __restrict__ out,    // [B][L][D]
                 float* __restrict__ attn)   // [B][L][L]
{
    extern __shared__ float sS[];           // [BM][L_] fp32 scores/probs, 256 KB

    const int b     = blockIdx.y;
    const int qBase = blockIdx.x * BM;
    const int tid   = threadIdx.x;
    const int wid   = tid >> 5;             // wave id 0..7 (wave32)
    const int lane  = tid & 31;
    const int lhalf = lane & 15;
    const int hi    = lane >> 4;            // 0 | 1

    const float* Qb = qu + (size_t)b * L_ * D_;
    const float* Kb = ke + (size_t)b * L_ * D_;
    const float* Vb = va + (size_t)b * L_ * D_;

    // ---------------- Phase 1: S = (Q K^T) * scale  ->  LDS ----------------
    // Preload Q A-fragments: 2 M-tiles x 4 K-chunks (D=128 = 4*32)
    v16h aQ[2][4];
    #pragma unroll
    for (int mt = 0; mt < 2; ++mt) {
        const float* qrow = Qb + (size_t)(qBase + mt * 16 + lhalf) * D_;
        #pragma unroll
        for (int c = 0; c < 4; ++c) {
            v16h a;
            #pragma unroll
            for (int v = 0; v < 8; ++v) {
                const int k = c * 32 + kOff(v, hi);
                float2 t = *(const float2*)(qrow + k);
                a[2 * v]     = (_Float16)t.x;
                a[2 * v + 1] = (_Float16)t.y;
            }
            aQ[mt][c] = a;
        }
    }

    // Each wave owns 16 of the 128 column-tiles of S
    for (int nt = wid; nt < L_ / 16; nt += NWAVES) {
        const float* krow = Kb + (size_t)(nt * 16 + lhalf) * D_;
        // hint the next K stripe this wave will touch into cache
        __builtin_prefetch(krow + (size_t)NWAVES * 16 * D_, 0, 1);

        v16h bf[4];
        #pragma unroll
        for (int c = 0; c < 4; ++c) {
            v16h bb;
            #pragma unroll
            for (int v = 0; v < 8; ++v) {
                const int k = c * 32 + kOff(v, hi);
                float2 t = *(const float2*)(krow + k);
                bb[2 * v]     = (_Float16)t.x;
                bb[2 * v + 1] = (_Float16)t.y;
            }
            bf[c] = bb;
        }

        v8f c0 = {}; v8f c1 = {};
        #pragma unroll
        for (int c = 0; c < 4; ++c) {
            c0 = __builtin_amdgcn_wmma_f32_16x16x32_f16(false, aQ[0][c],
                     false, bf[c], (short)0, c0, false, false);
            c1 = __builtin_amdgcn_wmma_f32_16x16x32_f16(false, aQ[1][c],
                     false, bf[c], (short)0, c1, false, false);
        }

        // C layout: dword v -> M = v + 8*hi, N = lane&15
        const int col = nt * 16 + lhalf;
        #pragma unroll
        for (int v = 0; v < 8; ++v) {
            const int m0 = v + 8 * hi;
            sS[(size_t)m0 * L_ + col]        = c0[v] * SCALE;
            sS[(size_t)(16 + m0) * L_ + col] = c1[v] * SCALE;
        }
    }
    __syncthreads();

    // ---------------- Phase 2: fp32 softmax in LDS + write attn ------------
    float* attnB = attn + ((size_t)b * L_ + qBase) * L_;
    for (int r = wid; r < BM; r += NWAVES) {       // one wave per row
        float* row = sS + (size_t)r * L_;
        float m = -3.4e38f;
        for (int i = lane; i < L_; i += 32) m = fmaxf(m, row[i]);
        #pragma unroll
        for (int off = 16; off > 0; off >>= 1) m = fmaxf(m, __shfl_xor(m, off, 32));

        float s = 0.0f;
        for (int i = lane; i < L_; i += 32) {
            float e = __expf(row[i] - m);
            row[i] = e;
            s += e;
        }
        #pragma unroll
        for (int off = 16; off > 0; off >>= 1) s += __shfl_xor(s, off, 32);

        const float inv = 1.0f / s;
        float* arow = attnB + (size_t)r * L_;
        for (int i = lane; i < L_; i += 32) {      // coalesced 128B stores
            float p = row[i] * inv;
            row[i]  = p;
            arow[i] = p;
        }
    }
    __syncthreads();

    // ---------------- Phase 3: O = P(32x2048) * V(2048x128) ----------------
    const int mt  = wid >> 2;           // waves 0-3 -> rows 0-15, 4-7 -> 16-31
    const int nt0 = wid & 3;            // two d-column tiles per wave
    const int nt1 = nt0 + 4;

    v8f o0 = {}; v8f o1 = {};
    const float* prow = sS + (size_t)(mt * 16 + lhalf) * L_;

    for (int kc = 0; kc < L_ / 32; ++kc) {
        v16h ap;                                    // A frag: P rows (LDS, fp32->f16)
        #pragma unroll
        for (int v = 0; v < 8; ++v) {
            const int k = kc * 32 + kOff(v, hi);
            float2 t = *(const float2*)(prow + k);
            ap[2 * v]     = (_Float16)t.x;
            ap[2 * v + 1] = (_Float16)t.y;
        }

        v16h b0, b1;                                // B frags: V columns
        #pragma unroll
        for (int v = 0; v < 8; ++v) {
            const int k = kc * 32 + kOff(v, hi);    // K pair = rows k, k+1 of V
            const float* vr = Vb + (size_t)k * D_;
            b0[2 * v]     = (_Float16)vr[nt0 * 16 + lhalf];
            b0[2 * v + 1] = (_Float16)vr[D_ + nt0 * 16 + lhalf];
            b1[2 * v]     = (_Float16)vr[nt1 * 16 + lhalf];
            b1[2 * v + 1] = (_Float16)vr[D_ + nt1 * 16 + lhalf];
        }

        o0 = __builtin_amdgcn_wmma_f32_16x16x32_f16(false, ap, false, b0,
                                                    (short)0, o0, false, false);
        o1 = __builtin_amdgcn_wmma_f32_16x16x32_f16(false, ap, false, b1,
                                                    (short)0, o1, false, false);
    }

    float* outB = out + ((size_t)b * L_ + qBase) * D_;
    #pragma unroll
    for (int v = 0; v < 8; ++v) {
        const int m = mt * 16 + v + 8 * hi;
        outB[(size_t)m * D_ + nt0 * 16 + lhalf] = o0[v];
        outB[(size_t)m * D_ + nt1 * 16 + lhalf] = o1[v];
    }
}

extern "C" void kernel_launch(void* const* d_in, const int* in_sizes, int n_in,
                              void* d_out, int out_size, void* d_ws, size_t ws_size,
                              hipStream_t stream) {
    (void)in_sizes; (void)n_in; (void)d_ws; (void)ws_size; (void)out_size;
    const float* qu = (const float*)d_in[0];
    const float* ke = (const float*)d_in[1];
    const float* va = (const float*)d_in[2];

    float* out  = (float*)d_out;                         // [B][L][D] first
    float* attn = out + (size_t)B_ * L_ * D_;            // then [B][L][L]

    dim3 grid(L_ / BM, B_);          // 64 q-tiles x 32 batches
    dim3 block(256);                 // 8 wave32s
    size_t ldsBytes = (size_t)BM * L_ * sizeof(float);   // 256 KB of the 320 KB WGP LDS
    sdpa_wmma_kernel<<<grid, block, ldsBytes, stream>>>(qu, ke, va, out, attn);
}
